// GroupQLinear_5746666242136
// MI455X (gfx1250) — compile-verified, matrix operand order checked by reading, not run on hardware
//
#include <hip/hip_runtime.h>
#include <hip/hip_bf16.h>

// ---------------------------------------------------------------------------
// GroupQLinear for MI455X (gfx1250, wave32, WMMA).
//   y = dequant(x_q) @ W^T + bias   -> group_quant(y)
// M = B*T = 8192, K = H = 4096, N = O = 4096 (derived from in_sizes).
// Kernel 1: bf16 WMMA GEMM (16x16x32). Block tile 128x256, 256 threads
//           (8 waves, 2x4 wave grid), 4x4 WMMA tiles per wave ->
//           16 ds_load_b128 per 16 v_wmma per k-step (1:1).
// Kernel 2: per-token requantization, in place on the y_q output region.
// ---------------------------------------------------------------------------

typedef __attribute__((ext_vector_type(16))) __bf16 v16bf;
typedef __attribute__((ext_vector_type(8)))  float  v8f;

#define BM   128
#define BN   256
#define BK   64
#define BKP  80              // padded row stride (halfs): 160B, multiple of 16B
#define QMAXF 127.0f

union FragU { v16bf v; float4 f[2]; };

__global__ __launch_bounds__(256)
void gqlin_gemm_bf16(const int*   __restrict__ xq,
                     const float* __restrict__ xdelta,
                     const int*   __restrict__ xe,
                     const float* __restrict__ w,
                     const float* __restrict__ bias,
                     float*       __restrict__ y,
                     int M, int H, int O, int GH) {
    __shared__ __bf16 As[BM][BKP];   // activations (dequantized), 20 KB
    __shared__ __bf16 Bs[BN][BKP];   // weights, 40 KB

    const int tid   = threadIdx.x;
    const int lane  = tid & 31;
    const int wave  = tid >> 5;          // 0..7
    const int waveM = wave >> 2;         // 0..1 -> 64-row slab
    const int waveN = wave & 3;          // 0..3 -> 64-col slab
    const int hl    = lane >> 4;         // half-wave select (K-halves)
    const int l16   = lane & 15;

    const int n0 = blockIdx.x * BN;
    const int m0 = blockIdx.y * BM;

    v8f acc[4][4];
    #pragma unroll
    for (int i = 0; i < 4; ++i)
        #pragma unroll
        for (int j = 0; j < 4; ++j)
            acc[i][j] = (v8f){0.f,0.f,0.f,0.f,0.f,0.f,0.f,0.f};

    for (int k0 = 0; k0 < H; k0 += BK) {
        // ---- stage A tile: dequantize int8 (stored int32) -> bf16 ----
        #pragma unroll
        for (int it = 0; it < (BM * (BK / 4)) / 256; ++it) {
            const int slot = tid + it * 256;          // 0..2047
            const int r    = slot >> 4;               // row in tile
            const int c4   = (slot & 15) << 2;        // col (x4)
            const int grow = m0 + r;
            const int4 q4 = *reinterpret_cast<const int4*>(
                xq + (size_t)grow * H + k0 + c4);
            const int   e  = xe[(size_t)grow * GH + ((k0 + c4) >> 5)];
            const float s  = __builtin_amdgcn_ldexpf(xdelta[grow], -e);
            As[r][c4 + 0] = (__bf16)((float)q4.x * s);
            As[r][c4 + 1] = (__bf16)((float)q4.y * s);
            As[r][c4 + 2] = (__bf16)((float)q4.z * s);
            As[r][c4 + 3] = (__bf16)((float)q4.w * s);
        }
        // ---- stage B tile: weight fp32 -> bf16 (W is [O,H] row-major) ----
        #pragma unroll
        for (int it = 0; it < (BN * (BK / 4)) / 256; ++it) {
            const int slot = tid + it * 256;          // 0..4095
            const int n    = slot >> 4;
            const int c4   = (slot & 15) << 2;
            const float4 wv = *reinterpret_cast<const float4*>(
                w + (size_t)(n0 + n) * H + k0 + c4);
            Bs[n][c4 + 0] = (__bf16)wv.x;
            Bs[n][c4 + 1] = (__bf16)wv.y;
            Bs[n][c4 + 2] = (__bf16)wv.z;
            Bs[n][c4 + 3] = (__bf16)wv.w;
        }
        __syncthreads();

        // ---- two K=32 WMMA steps over the 64-wide tile ----
        #pragma unroll
        for (int ks = 0; ks < BK; ks += 32) {
            // 16-bit A/B fragment layout (ISA 7.12.2): lane half 0 holds
            // K 0..7 / 16..23, half 1 holds K 8..15 / 24..31 -> 2x b128 each.
            FragU a[4], b[4];
            #pragma unroll
            for (int i = 0; i < 4; ++i) {
                const int row = waveM * 64 + i * 16 + l16;
                a[i].f[0] = *reinterpret_cast<const float4*>(&As[row][ks + hl * 8]);
                a[i].f[1] = *reinterpret_cast<const float4*>(&As[row][ks + 16 + hl * 8]);
            }
            #pragma unroll
            for (int j = 0; j < 4; ++j) {
                const int row = waveN * 64 + j * 16 + l16;
                b[j].f[0] = *reinterpret_cast<const float4*>(&Bs[row][ks + hl * 8]);
                b[j].f[1] = *reinterpret_cast<const float4*>(&Bs[row][ks + 16 + hl * 8]);
            }
            #pragma unroll
            for (int i = 0; i < 4; ++i)
                #pragma unroll
                for (int j = 0; j < 4; ++j)
                    acc[i][j] = __builtin_amdgcn_wmma_f32_16x16x32_bf16(
                        false, a[i].v, false, b[j].v,
                        (short)0, acc[i][j], false, false);
        }
        __syncthreads();
    }

    // ---- epilogue: + bias, write fp32 y (D layout: lane half -> M+8) ----
    #pragma unroll
    for (int j = 0; j < 4; ++j) {
        const int gcol = n0 + waveN * 64 + j * 16 + l16;
        const float bv = bias[gcol];
        #pragma unroll
        for (int i = 0; i < 4; ++i) {
            const int rbase = m0 + waveM * 64 + i * 16 + hl * 8;
            #pragma unroll
            for (int v = 0; v < 8; ++v)
                y[(size_t)(rbase + v) * O + gcol] = acc[i][j][v] + bv;
        }
    }
}

// ---------------------------------------------------------------------------
// Kernel 2: group_quant of y, one block per token. y is quantized in place
// (d_out stores y_q as float per harness convention), y_delta and y_e are
// written to their regions of d_out.
// ---------------------------------------------------------------------------
__global__ __launch_bounds__(256)
void gqlin_requant(float* __restrict__ y,
                   float* __restrict__ y_delta,
                   float* __restrict__ y_e,
                   int O, int OG) {
    __shared__ float red[256];
    const int tok  = blockIdx.x;
    const int tid  = threadIdx.x;
    const size_t base = (size_t)tok * O;

    // per-token absmax
    float amax = 0.f;
    for (int c = tid; c < O; c += 256)
        amax = fmaxf(amax, fabsf(y[base + c]));
    red[tid] = amax;
    __syncthreads();
    #pragma unroll
    for (int s = 128; s > 0; s >>= 1) {
        if (tid < s) red[tid] = fmaxf(red[tid], red[tid + s]);
        __syncthreads();
    }
    const float absmax = red[0];
    const float delta  = absmax / QMAXF;
    if (tid == 0) y_delta[tok] = delta;
    __syncthreads();

    // one thread per group of 32
    if (tid < OG) {
        const int g = tid;
        float vals[32];
        float gmax = 0.f;
        #pragma unroll
        for (int q = 0; q < 8; ++q) {
            const float4 v4 = *reinterpret_cast<const float4*>(&y[base + g * 32 + q * 4]);
            vals[q * 4 + 0] = v4.x; vals[q * 4 + 1] = v4.y;
            vals[q * 4 + 2] = v4.z; vals[q * 4 + 3] = v4.w;
            gmax = fmaxf(gmax, fmaxf(fmaxf(fabsf(v4.x), fabsf(v4.y)),
                                     fmaxf(fabsf(v4.z), fabsf(v4.w))));
        }
        const float ratio = absmax / fmaxf(gmax, 1e-8f);
        const float e     = fminf(fmaxf(floorf(log2f(fmaxf(ratio, 1.f))), 0.f), 15.f);
        y_e[(size_t)tok * OG + g] = e;
        const float scale = __builtin_amdgcn_ldexpf(delta, -(int)e);
        const float inv   = 1.f / fmaxf(scale, 1e-8f);
        #pragma unroll
        for (int c = 0; c < 32; ++c) {
            const float q = fminf(fmaxf(rintf(vals[c] * inv), -QMAXF), QMAXF);
            y[base + g * 32 + c] = q;
        }
    }
}

extern "C" void kernel_launch(void* const* d_in, const int* in_sizes, int n_in,
                              void* d_out, int out_size, void* d_ws, size_t ws_size,
                              hipStream_t stream) {
    const int*   xq     = (const int*)  d_in[0];
    const float* xdelta = (const float*)d_in[1];
    const int*   xe     = (const int*)  d_in[2];
    const float* weight = (const float*)d_in[3];
    const float* bias   = (const float*)d_in[4];

    const int M  = in_sizes[1];            // B*T tokens
    const int H  = in_sizes[0] / M;        // hidden
    const int O  = in_sizes[4];            // output features
    const int GH = in_sizes[2] / M;        // H/32 groups (input)
    const int OG = O / 32;                 // output groups per token

    float* y       = (float*)d_out;                         // y_q region (fp32 staged)
    float* y_delta = y + (size_t)M * O;                     // M floats
    float* y_e     = y_delta + M;                           // M*OG floats

    dim3 grid1(O / BN, M / BM);
    gqlin_gemm_bf16<<<grid1, 256, 0, stream>>>(xq, xdelta, xe, weight, bias,
                                               y, M, H, O, GH);
    gqlin_requant<<<M, 256, 0, stream>>>(y, y_delta, y_e, O, OG);
}
